// CrossCycleSelfAttention_69260642615587
// MI455X (gfx1250) — matrix operand order repaired and checked by reading, not compiled, not run on hardware
//
#include <hip/hip_runtime.h>

// Problem constants
#define B_ 16
#define C_ 16
#define P_ 512
#define D_ 256

typedef __bf16 bf16;
typedef __attribute__((ext_vector_type(16))) __bf16 bf16x16;
typedef __attribute__((ext_vector_type(8)))  __bf16 bf16x8;
typedef __attribute__((ext_vector_type(8)))  float  f32x8;
typedef __attribute__((ext_vector_type(4)))  int    v4i;

// Native f32 -> bf16 (should lower to v_cvt_pk_bf16_f32-class hardware cvt)
__device__ __forceinline__ bf16 f2bf(float f) { return (bf16)f; }
__device__ __forceinline__ unsigned pkbf(float lo, float hi) {
  unsigned short a = __builtin_bit_cast(unsigned short, (bf16)lo);
  unsigned short b = __builtin_bit_cast(unsigned short, (bf16)hi);
  return (unsigned)a | ((unsigned)b << 16);
}

__device__ __forceinline__ bf16x16 join16(bf16x8 lo, bf16x8 hi) {
  return __builtin_shufflevector(lo, hi, 0,1,2,3,4,5,6,7,8,9,10,11,12,13,14,15);
}

#define DS_WAIT() asm volatile("s_wait_dscnt 0x0" ::: "memory")

#if __has_builtin(__builtin_amdgcn_global_load_async_to_lds_b128)
#define USE_ASYNC 1
#if __has_builtin(__builtin_amdgcn_s_wait_asynccnt)
#define ASYNC_WAIT() __builtin_amdgcn_s_wait_asynccnt(0)
#else
#define ASYNC_WAIT() asm volatile("s_wait_asynccnt 0x0" ::: "memory")
#endif
#endif

// WT (W^T bf16 tile) row stride: padded for LDS bank spread, 80B keeps b128 align
#define WT_STRIDE 40

// ---------------------------------------------------------------------------
// Kernel 1: S[b,p,d] = sum_c aw[c,p,d] * query[b,c,p,d]   (context = S - aw*q)
// ---------------------------------------------------------------------------
__global__ void __launch_bounds__(256) ctx_sum_kernel(
    const float* __restrict__ query, const float* __restrict__ aw,
    float* __restrict__ S) {
  const long long t  = (long long)blockIdx.x * blockDim.x + threadIdx.x;
  const long long o  = t * 4;                       // one float4 per thread
  const int b  = (int)(o / (P_ * D_));
  const int pd = (int)(o % (P_ * D_));
  float4 acc = make_float4(0.f, 0.f, 0.f, 0.f);
#pragma unroll
  for (int c = 0; c < C_; ++c) {
    float4 qv = *(const float4*)(query + (size_t)b * C_ * P_ * D_ +
                                 (size_t)c * P_ * D_ + pd);
    float4 wv = *(const float4*)(aw + (size_t)c * P_ * D_ + pd);
    acc.x += qv.x * wv.x; acc.y += qv.y * wv.y;
    acc.z += qv.z * wv.z; acc.w += qv.w * wv.w;
  }
  *(float4*)(S + (size_t)b * P_ * D_ + pd) = acc;
}

// ---------------------------------------------------------------------------
// Kernel 2: per (b,c): Q = relu(q@Wq+bq)*D^-1/2, K = relu(ctx@Wk+bk),
//           V = relu(ctx@Wv+bv) (stored transposed [D][P]) -- all bf16.
// grid: (B*C)*(P/64), 256 threads = 8 waves; wave -> 16 rows x 128 cols
// LDS: [0,20480)  WT      : W^T chunk bf16 [256 e][40 stride]
//      [20480,..) Wraw    : raw f32 W chunk (async staging)   \ aliased
//      [20480,..) OUTS    : per-wave 16x128 bf16 out staging  / (time-sliced)
// ---------------------------------------------------------------------------
__global__ void __launch_bounds__(256) proj_kernel(
    const float* __restrict__ query, const float* __restrict__ aw,
    const float* __restrict__ Wq, const float* __restrict__ Wk,
    const float* __restrict__ Wv, const float* __restrict__ bq,
    const float* __restrict__ bk, const float* __restrict__ bv,
    const float* __restrict__ S,
    bf16* __restrict__ Qw, bf16* __restrict__ Kw, bf16* __restrict__ VTw) {
  __shared__ __align__(16) char smem[20480 + 34816];
  bf16*  WT   = (bf16*)smem;                 // 256*40*2 = 20480 B
  float* Wraw = (float*)(smem + 20480);      // 32*256*4 = 32768 B
  bf16*  OUTS = (bf16*)(smem + 20480);       // 8*16*136*2 = 34816 B

  const int tid   = threadIdx.x;
  const int wave  = tid >> 5;
  const int lane  = tid & 31;
  const int n     = lane & 15;
  const int half  = lane >> 4;

  const int blk   = blockIdx.x;
  const int ptile = blk & 7;                 // P_/64 = 8 tiles
  const int bc    = blk >> 3;
  const int b     = bc >> 4;                 // bc / C_
  const int c     = bc & (C_ - 1);
  const int mbase = (wave & 3) * 16;
  const int ebase = (wave >> 2) * 128;
  const int prow  = ptile * 64 + mbase + n;  // A-fragment row (M = lane%16)

  const size_t qoff = ((size_t)bc * P_ + prow) * D_;
  const size_t soff = ((size_t)b * P_ + prow) * D_;
  const size_t aoff = ((size_t)c * P_ + prow) * D_;

  const int eg = tid & 63;                   // transpose helper: e group of 4
  const int dp = tid >> 6;                   // d-pair slot

  for (int proj = 0; proj < 3; ++proj) {
    const float* W  = (proj == 0) ? Wq : (proj == 1) ? Wk : Wv;
    const float* bb = (proj == 0) ? bq : (proj == 1) ? bk : bv;
    f32x8 acc[8] = {};

    for (int kk = 0; kk < D_; kk += 32) {
      const float* gsrc = W + (size_t)c * D_ * D_ + (size_t)kk * D_;  // 32x256 f32, contiguous
      __syncthreads();
#ifdef USE_ASYNC
      // async DMA the raw f32 chunk into LDS (ASYNCcnt-tracked)
#pragma unroll
      for (int i = 0; i < 8; ++i) {
        const int ch = i * 256 + tid;        // 16B chunk id, 2048 total
        __builtin_amdgcn_global_load_async_to_lds_b128(
            (v4i*)(gsrc + ch * 4), (v4i*)(Wraw + ch * 4), 0, 0);
      }
      ASYNC_WAIT();
      __syncthreads();
#endif
      // transpose + f32->bf16 into WT[e][dlocal] (padded stride)
#pragma unroll
      for (int i = 0; i < 4; ++i) {
        const int d0 = i * 8 + dp * 2;
#ifdef USE_ASYNC
        float4 r0 = *(const float4*)(Wraw + d0 * 256 + eg * 4);
        float4 r1 = *(const float4*)(Wraw + (d0 + 1) * 256 + eg * 4);
#else
        float4 r0 = *(const float4*)(gsrc + d0 * 256 + eg * 4);
        float4 r1 = *(const float4*)(gsrc + (d0 + 1) * 256 + eg * 4);
#endif
        *(unsigned*)&WT[(eg * 4 + 0) * WT_STRIDE + d0] = pkbf(r0.x, r1.x);
        *(unsigned*)&WT[(eg * 4 + 1) * WT_STRIDE + d0] = pkbf(r0.y, r1.y);
        *(unsigned*)&WT[(eg * 4 + 2) * WT_STRIDE + d0] = pkbf(r0.z, r1.z);
        *(unsigned*)&WT[(eg * 4 + 3) * WT_STRIDE + d0] = pkbf(r0.w, r1.w);
      }
      __syncthreads();

      // A fragment: 16x32 bf16; a[0..7]=K kk+half*8.., a[8..15]=K kk+16+half*8..
      float af[16];
      if (proj == 0) {
        const float4* p0 = (const float4*)(query + qoff + kk + half * 8);
        const float4* p1 = (const float4*)(query + qoff + kk + 16 + half * 8);
        float4 x0 = p0[0], x1 = p0[1], y0 = p1[0], y1 = p1[1];
        af[0]=x0.x; af[1]=x0.y; af[2]=x0.z; af[3]=x0.w;
        af[4]=x1.x; af[5]=x1.y; af[6]=x1.z; af[7]=x1.w;
        af[8]=y0.x; af[9]=y0.y; af[10]=y0.z; af[11]=y0.w;
        af[12]=y1.x; af[13]=y1.y; af[14]=y1.z; af[15]=y1.w;
      } else {
        // context = S - aw*query
#pragma unroll
        for (int g = 0; g < 2; ++g) {
          const int dbase = kk + g * 16 + half * 8;
          const float4* ps = (const float4*)(S + soff + dbase);
          const float4* pa = (const float4*)(aw + aoff + dbase);
          const float4* pq = (const float4*)(query + qoff + dbase);
#pragma unroll
          for (int v = 0; v < 2; ++v) {
            float4 sv = ps[v], av = pa[v], qv = pq[v];
            af[g*8 + v*4 + 0] = sv.x - av.x * qv.x;
            af[g*8 + v*4 + 1] = sv.y - av.y * qv.y;
            af[g*8 + v*4 + 2] = sv.z - av.z * qv.z;
            af[g*8 + v*4 + 3] = sv.w - av.w * qv.w;
          }
        }
      }
      bf16x16 a;
#pragma unroll
      for (int i = 0; i < 16; ++i) a[i] = f2bf(af[i]);

#pragma unroll
      for (int t = 0; t < 8; ++t) {
        // B fragment: lane column e = ebase+t*16+n; K = half*8 + i, 16 + half*8 + i
        const bf16* wp = &WT[(ebase + t * 16 + n) * WT_STRIDE + half * 8];
        bf16x16 bfr = join16(*(const bf16x8*)wp, *(const bf16x8*)(wp + 16));
        acc[t] = __builtin_amdgcn_wmma_f32_16x16x32_bf16(
            false, a, false, bfr, (short)0, acc[t], false, false);
      }
    }

    // Epilogue: C/D layout -> row = mbase_g + r + 8*half, col = ebase+t*16+n
    const int rowb = ptile * 64 + mbase + 8 * half;
    if (proj == 2) {
      // V transposed: VT[bc][e][p]; rows r contiguous -> packed b128 stores
#pragma unroll
      for (int t = 0; t < 8; ++t) {
        const int e = ebase + t * 16 + n;
        const float bias = bb[(size_t)c * D_ + e];
        unsigned pk[4];
#pragma unroll
        for (int r = 0; r < 4; ++r)
          pk[r] = pkbf(fmaxf(acc[t][2*r]   + bias, 0.f),
                       fmaxf(acc[t][2*r+1] + bias, 0.f));
        *reinterpret_cast<uint4*>(VTw + ((size_t)bc * D_ + e) * P_ + rowb) =
            make_uint4(pk[0], pk[1], pk[2], pk[3]);
      }
    } else {
      // stage wave-private 16x128 bf16 tile in LDS, read back coalesced
      bf16* os = OUTS + wave * 16 * 136;
      const float sc = (proj == 0) ? 0.0625f : 1.0f;    // D^-0.5 = 1/16
#pragma unroll
      for (int t = 0; t < 8; ++t) {
        const int e = ebase + t * 16 + n;
        const float bias = bb[(size_t)c * D_ + e];
#pragma unroll
        for (int r = 0; r < 8; ++r)
          os[(r + 8 * half) * 136 + t * 16 + n] =
              f2bf(fmaxf(acc[t][r] + bias, 0.f) * sc);
      }
      DS_WAIT();
      bf16* dst = (proj == 0) ? Qw : Kw;
#pragma unroll
      for (int i = 0; i < 8; ++i) {
        const int idx = lane + 32 * i;      // 16 rows x 16 chunks of 8 bf16
        const int row = idx & 15, ch = idx >> 4;
        uint4 v = *(const uint4*)(os + row * 136 + ch * 8);
        *reinterpret_cast<uint4*>(
            dst + ((size_t)bc * P_ + ptile * 64 + mbase + row) * D_ +
            ebase + ch * 8) = v;
      }
      DS_WAIT();                            // drain before staging aliases OUTS
    }
  }
}

// ---------------------------------------------------------------------------
// Kernel 3: flash attention per (b,c, 128-row q tile). 8 waves x 16 rows.
// scores = Q@K^T (bf16 WMMA), online softmax, O += P@V (V transposed).
// ---------------------------------------------------------------------------
__global__ void __launch_bounds__(256) attn_kernel(
    const bf16* __restrict__ Qw, const bf16* __restrict__ Kw,
    const bf16* __restrict__ VTw, float* __restrict__ out) {
  __shared__ __align__(16) bf16 Plds[8 * 16 * 136];  // per-wave 16x128 (+pad)

  const int tid  = threadIdx.x;
  const int wave = tid >> 5;
  const int lane = tid & 31;
  const int n    = lane & 15;
  const int half = lane >> 4;

  const int blk   = blockIdx.x;
  const int qt    = blk & 3;                 // P_/128 = 4 q tiles
  const int bc    = blk >> 2;
  const int qbase = qt * 128 + wave * 16;

  const bf16* Qp = Qw  + (size_t)bc * P_ * D_;
  const bf16* Kp = Kw  + (size_t)bc * P_ * D_;
  const bf16* Vp = VTw + (size_t)bc * D_ * P_;
  bf16* Pw = &Plds[wave * 16 * 136];

  f32x8 Oacc[16] = {};
  float mrow[8], lrow[8];
#pragma unroll
  for (int r = 0; r < 8; ++r) { mrow[r] = -1e30f; lrow[r] = 0.f; }

  for (int jt = 0; jt < 4; ++jt) {
    const int j0 = jt * 128;

    // prefetch next KV tile into cache (global_prefetch_b8)
    if (jt < 3) {
      __builtin_prefetch(Kp + (size_t)(j0 + 128 + (tid & 127)) * D_, 0, 1);
      __builtin_prefetch(Vp + (size_t)tid * P_ + j0 + 128, 0, 1);
    }

    // ---- scores: 16 x 128 per wave ----
    f32x8 Sacc[8] = {};
    for (int kk = 0; kk < D_; kk += 32) {
      const bf16* ap = Qp + (size_t)(qbase + n) * D_ + kk;
      bf16x16 a = join16(*(const bf16x8*)(ap + half * 8),
                         *(const bf16x8*)(ap + 16 + half * 8));
#pragma unroll
      for (int t = 0; t < 8; ++t) {
        const bf16* bp = Kp + (size_t)(j0 + t * 16 + n) * D_ + kk + half * 8;
        bf16x16 bfr = join16(*(const bf16x8*)bp, *(const bf16x8*)(bp + 16));
        Sacc[t] = __builtin_amdgcn_wmma_f32_16x16x32_bf16(
            false, a, false, bfr, (short)0, Sacc[t], false, false);
      }
    }

    // ---- online softmax (rows live on 16-lane halves; shfl_xor reduce) ----
    float mt[8], scale[8];
#pragma unroll
    for (int r = 0; r < 8; ++r) {
      float m = -1e30f;
#pragma unroll
      for (int t = 0; t < 8; ++t) m = fmaxf(m, Sacc[t][r]);
#pragma unroll
      for (int s = 1; s < 16; s <<= 1) m = fmaxf(m, __shfl_xor(m, s, 32));
      mt[r] = fmaxf(m, mrow[r]);
    }
#pragma unroll
    for (int r = 0; r < 8; ++r) {
      scale[r] = __expf(mrow[r] - mt[r]);
      float s = 0.f;
#pragma unroll
      for (int t = 0; t < 8; ++t) {
        float p = __expf(Sacc[t][r] - mt[r]);
        Sacc[t][r] = p;
        s += p;
      }
#pragma unroll
      for (int sh = 1; sh < 16; sh <<= 1) s += __shfl_xor(s, sh, 32);
      lrow[r] = lrow[r] * scale[r] + s;
      mrow[r] = mt[r];
    }

    // ---- P -> LDS (bf16, per-wave private region; no barrier needed) ----
#pragma unroll
    for (int t = 0; t < 8; ++t)
#pragma unroll
      for (int r = 0; r < 8; ++r)
        Pw[(r + 8 * half) * 136 + t * 16 + n] = f2bf(Sacc[t][r]);

    // rescale O while DS stores drain
#pragma unroll
    for (int u = 0; u < 16; ++u)
#pragma unroll
      for (int r = 0; r < 8; ++r) Oacc[u][r] *= scale[r];

    DS_WAIT();                               // cross-lane RAW through LDS

    // ---- O += P @ V ----
    for (int jj = 0; jj < 128; jj += 32) {
      const bf16* ap = Pw + n * 136 + jj;
      bf16x16 a = join16(*(const bf16x8*)(ap + half * 8),
                         *(const bf16x8*)(ap + 16 + half * 8));
#pragma unroll
      for (int u = 0; u < 16; ++u) {
        const bf16* bp = Vp + (size_t)(u * 16 + n) * P_ + j0 + jj + half * 8;
        bf16x16 bfr = join16(*(const bf16x8*)bp, *(const bf16x8*)(bp + 16));
        Oacc[u] = __builtin_amdgcn_wmma_f32_16x16x32_bf16(
            false, a, false, bfr, (short)0, Oacc[u], false, false);
      }
    }
  }

  // ---- normalize + store via LDS transpose (coalesced b128 f32 stores) ----
  float rinv[8];
#pragma unroll
  for (int r = 0; r < 8; ++r) rinv[r] = 1.0f / lrow[r];
  float* Pf = reinterpret_cast<float*>(Pw);   // 16 x 68 f32 region (4352 B)
  for (int qn = 0; qn < 4; ++qn) {            // 4 quarters of E=256
#pragma unroll
    for (int uu = 0; uu < 4; ++uu) {
      const int u = qn * 4 + uu;
#pragma unroll
      for (int r = 0; r < 8; ++r)
        Pf[(r + 8 * half) * 68 + uu * 16 + n] = Oacc[u][r] * rinv[r];
    }
    DS_WAIT();
#pragma unroll
    for (int i = 0; i < 8; ++i) {
      const int idx = lane + 32 * i;          // 16 rows x 16 float4 chunks
      const int row = idx & 15, ch = idx >> 4;
      float4 v = *(const float4*)(Pf + row * 68 + ch * 4);
      *(float4*)(out + ((size_t)bc * P_ + qbase + row) * D_ + qn * 64 + ch * 4) = v;
    }
    DS_WAIT();                                // WAR before next quarter reuse
  }
}

// ---------------------------------------------------------------------------
// Workspace layout (bytes):
//   S  : B*P*D*4           =   8,388,608   @ 0
//   Qw : B*C*P*D*2         =  67,108,864   @ 8,388,608
//   Kw : B*C*P*D*2         =  67,108,864   @ 75,497,472
//   VTw: B*C*D*P*2         =  67,108,864   @ 142,606,336
// total ~200 MB (requires ws_size >= 209,715,200)
// ---------------------------------------------------------------------------
extern "C" void kernel_launch(void* const* d_in, const int* in_sizes, int n_in,
                              void* d_out, int out_size, void* d_ws,
                              size_t ws_size, hipStream_t stream) {
  const float* query = (const float*)d_in[0];
  const float* aw    = (const float*)d_in[1];
  const float* Wq    = (const float*)d_in[2];
  const float* Wk    = (const float*)d_in[3];
  const float* Wv    = (const float*)d_in[4];
  const float* bq    = (const float*)d_in[5];
  const float* bk    = (const float*)d_in[6];
  const float* bv    = (const float*)d_in[7];
  float* out = (float*)d_out;

  char* ws = (char*)d_ws;
  float* S  = (float*)ws;
  bf16* Qw  = (bf16*)(ws + 8388608ull);
  bf16* Kw  = (bf16*)(ws + 75497472ull);
  bf16* VTw = (bf16*)(ws + 142606336ull);

  ctx_sum_kernel<<<(B_ * P_ * D_ / 4 + 255) / 256, 256, 0, stream>>>(query, aw, S);
  proj_kernel<<<B_ * C_ * (P_ / 64), 256, 0, stream>>>(
      query, aw, Wq, Wk, Wv, bq, bk, bv, S, Qw, Kw, VTw);
  attn_kernel<<<B_ * C_ * (P_ / 128), 256, 0, stream>>>(Qw, Kw, VTw, out);
}
